// AutoregMeshModel_5179730559549
// MI455X (gfx1250) — matrix operand order repaired and checked by reading, not compiled, not run on hardware
//
#include <hip/hip_runtime.h>
#include <hip/hip_bf16.h>

// ---------- types ----------
typedef _Float16 h8_t  __attribute__((ext_vector_type(8)));
typedef _Float16 h16_t __attribute__((ext_vector_type(16)));
typedef float    f8_t  __attribute__((ext_vector_type(8)));

__device__ __forceinline__ h16_t hcat(h8_t a, h8_t b) {
    h16_t r;
#pragma unroll
    for (int i = 0; i < 8; ++i) { r[i] = a[i]; r[i + 8] = b[i]; }
    return r;
}

__device__ __forceinline__ f8_t wmma_f32_f16(h16_t a, h16_t b, f8_t c) {
    // D = A(16x32 f16) x B(32x16 f16) + C(16x16 f32)
    return __builtin_amdgcn_wmma_f32_16x16x32_f16(false, a, false, b, (short)0, c,
                                                  false, false);
}

// ---------- constants ----------
#define NB     2
#define TQ     1024
#define TK     2048
#define DIM    1536
#define NH     16
#define HDIM   96

// ---------- fp32 -> fp16 convert ----------
__global__ void cvt_f16_kernel(const float* __restrict__ in, _Float16* __restrict__ out, int n) {
    int i = blockIdx.x * blockDim.x + threadIdx.x;
    if (i < n) out[i] = (_Float16)in[i];
}

// ---------- generic WMMA GEMM: C[M,N] = A[M,K] (f16) * W[N,K]^T (f16) + bias ----------
// Block: 256 thr = 8 waves = 2 m-waves x 4 n-waves; wave tile 16m x 128n (8 WMMA tiles)
// MODE 0: store f16 head-major  out[((b*NH+h)*T + t)*96 + hd]
// MODE 1: store f16 transposed  out[((b*NH+h)*96 + hd)*T + t]   (packed h8 stores)
// MODE 2: store f32 row-major   out[m*N + n] (+bias)  -> final output
template <int MODE>
__global__ void __launch_bounds__(256)
gemm_wmma_kernel(const _Float16* __restrict__ A, const _Float16* __restrict__ W,
                 const float* __restrict__ bias, void* __restrict__ outp,
                 int M, int N, int K, int T) {
    const int lane   = threadIdx.x & 31;
    const int wave   = threadIdx.x >> 5;
    const int lane16 = lane & 15;
    const bool hiH   = lane >= 16;
    const int wm = wave & 1;   // 2 m-waves
    const int wn = wave >> 1;  // 4 n-waves
    const int m0 = blockIdx.x * 32 + wm * 16;
    const int n0 = blockIdx.y * 512 + wn * 128;

    f8_t acc[8];
#pragma unroll
    for (int j = 0; j < 8; ++j)
#pragma unroll
        for (int r = 0; r < 8; ++r) acc[j][r] = 0.0f;

    // A fragment base: row = m0 + lane16; lanes>=16 start at K+16 (contiguous K, matches B)
    const _Float16* pa = A + (size_t)(m0 + lane16) * K + (hiH ? 16 : 0);

    for (int k0 = 0; k0 < K; k0 += 32) {
        if (k0 + 32 < K) __builtin_prefetch(pa + k0 + 32, 0, 3);
        h8_t alo = *(const h8_t*)(pa + k0);        // K = k0 + {0..7 | 16..23}
        h8_t ahi = *(const h8_t*)(pa + k0 + 8);    // K = k0 + {8..15 | 24..31}
        h16_t af = hcat(alo, ahi);
#pragma unroll
        for (int j = 0; j < 8; ++j) {
            // B column n = W row n (contiguous K); lanes>=16 hold K=k0+16..31
            const _Float16* pb =
                W + (size_t)(n0 + j * 16 + lane16) * K + k0 + (hiH ? 16 : 0);
            h16_t bf = *(const h16_t*)pb;
            acc[j] = wmma_f32_f16(af, bf, acc[j]);
        }
    }

    // C layout: lane L holds column N = L%16; VGPR r holds M = r (+8 for lanes>=16)
    if (MODE == 2) {
        float* out = (float*)outp;
#pragma unroll
        for (int j = 0; j < 8; ++j) {
            int n = n0 + j * 16 + lane16;
            float bv = bias[n];
#pragma unroll
            for (int r = 0; r < 8; ++r) {
                int m = m0 + r + (hiH ? 8 : 0);
                out[(size_t)m * N + n] = acc[j][r] + bv;
            }
        }
    } else if (MODE == 0) {
        _Float16* out = (_Float16*)outp;
#pragma unroll
        for (int j = 0; j < 8; ++j) {
            int n = n0 + j * 16 + lane16;
            int hh = n / HDIM, hd = n % HDIM;
            float bv = bias[n];
#pragma unroll
            for (int r = 0; r < 8; ++r) {
                int m = m0 + r + (hiH ? 8 : 0);
                int bb = m / T, t = m % T;
                out[(((size_t)bb * NH + hh) * T + t) * HDIM + hd] =
                    (_Float16)(acc[j][r] + bv);
            }
        }
    } else {  // MODE 1: transposed V; 8 consecutive t per lane -> one 16B store
        _Float16* out = (_Float16*)outp;
        int mb = m0 + (hiH ? 8 : 0);
        int bb = mb / T, t0 = mb % T;
#pragma unroll
        for (int j = 0; j < 8; ++j) {
            int n = n0 + j * 16 + lane16;
            int hh = n / HDIM, hd = n % HDIM;
            float bv = bias[n];
            h8_t v;
#pragma unroll
            for (int r = 0; r < 8; ++r) v[r] = (_Float16)(acc[j][r] + bv);
            *(h8_t*)(out + (((size_t)bb * NH + hh) * HDIM + hd) * (size_t)T + t0) = v;
        }
    }
}

// ---------- RoPE (in-place on head-major f16 buffer) ----------
__global__ void rope_kernel(_Float16* __restrict__ buf, const float* __restrict__ coords,
                            int T, int n) {
    int i = blockIdx.x * blockDim.x + threadIdx.x;
    if (i >= n) return;  // n = B*NH*T*48
    int f48 = i % 48;
    int t   = (i / 48) % T;
    int h   = (i / (48 * T)) % NH;
    int b   = i / (48 * T * NH);
    int f    = f48 & 15;
    int axis = f48 >> 4;
    // inv_freq = 10000^(-f/16)
    float invf = __expf(-((float)f * (1.0f / 16.0f)) * 9.210340371976184f);
    float ang  = coords[((size_t)b * T + t) * 3 + axis] * invf;
    float s = __sinf(ang), c = __cosf(ang);
    size_t base = (((size_t)b * NH + h) * T + t) * HDIM + f48;
    float x1 = (float)buf[base];
    float x2 = (float)buf[base + 48];
    buf[base]      = (_Float16)(x1 * c - x2 * s);
    buf[base + 48] = (_Float16)(x2 * c + x1 * s);
}

// ---------- flash attention: one wave per 16-query tile ----------
__global__ void __launch_bounds__(256)
attn_kernel(const _Float16* __restrict__ Qh, const _Float16* __restrict__ Kh,
            const _Float16* __restrict__ Vt, _Float16* __restrict__ AO) {
    __shared__ alignas(64) _Float16 pbuf[8][16][32];  // per-wave P tile (16q x 32k)

    const int lane   = threadIdx.x & 31;
    const int wave   = threadIdx.x >> 5;
    const int lane16 = lane & 15;
    const bool hiH   = lane >= 16;

    const int wgid   = blockIdx.x * 8 + wave;
    const int qtiles = TQ >> 4;                 // 64
    const int qt = wgid % qtiles;
    const int h  = (wgid / qtiles) % NH;
    const int b  = wgid / (qtiles * NH);
    const int q0 = qt * 16;

    const _Float16* Qb = Qh + (((size_t)b * NH + h) * TQ + q0) * HDIM;
    const _Float16* Kb = Kh + (((size_t)b * NH + h) * TK) * HDIM;
    const _Float16* Vb = Vt + (((size_t)b * NH + h) * HDIM) * (size_t)TK;

    const _Float16 hscale = (_Float16)0.1020620726159658f;  // 1/sqrt(96), folded into Q

    // Q A-fragments: 96 = 3 k-slices of 32; pre-scaled by 1/sqrt(Hd)
    h16_t qf[3];
#pragma unroll
    for (int s = 0; s < 3; ++s) {
        const _Float16* p = Qb + (size_t)lane16 * HDIM + s * 32 + (hiH ? 16 : 0);
        h8_t lo = *(const h8_t*)p;
        h8_t hi = *(const h8_t*)(p + 8);
#pragma unroll
        for (int i = 0; i < 8; ++i) { qf[s][i] = lo[i] * hscale; qf[s][i + 8] = hi[i] * hscale; }
    }

    // all-ones B fragment: O6 accumulates row sums (every lane sees its rows' sum)
    h16_t onesf;
#pragma unroll
    for (int i = 0; i < 16; ++i) onesf[i] = (_Float16)1.0f;

    f8_t o[6], o6;
    float m_run[8];
#pragma unroll
    for (int nt = 0; nt < 6; ++nt)
#pragma unroll
        for (int r = 0; r < 8; ++r) o[nt][r] = 0.0f;
#pragma unroll
    for (int r = 0; r < 8; ++r) { o6[r] = 0.0f; m_run[r] = -1e30f; }

    for (int kb = 0; kb < TK; kb += 32) {
        // S = (Q/sqrt(d)) K^T : two 16x16 C-tiles
        f8_t s0, s1;
#pragma unroll
        for (int r = 0; r < 8; ++r) { s0[r] = 0.0f; s1[r] = 0.0f; }
#pragma unroll
        for (int s = 0; s < 3; ++s) {
            const _Float16* pk0 =
                Kb + (size_t)(kb + lane16) * HDIM + s * 32 + (hiH ? 16 : 0);
            s0 = wmma_f32_f16(qf[s], *(const h16_t*)pk0, s0);
            const _Float16* pk1 =
                Kb + (size_t)(kb + 16 + lane16) * HDIM + s * 32 + (hiH ? 16 : 0);
            s1 = wmma_f32_f16(qf[s], *(const h16_t*)pk1, s1);
        }

        // online max; row m = r + 8*(lane>=16) lives across the 16-lane half
        float p0[8], p1[8], alpha[8];
#pragma unroll
        for (int r = 0; r < 8; ++r) {
            float mx = fmaxf(s0[r], s1[r]);
            for (int msk = 1; msk < 16; msk <<= 1)
                mx = fmaxf(mx, __shfl_xor(mx, msk, 32));
            float mnew = fmaxf(m_run[r], mx);
            alpha[r] = __expf(m_run[r] - mnew);
            p0[r] = __expf(s0[r] - mnew);
            p1[r] = __expf(s1[r] - mnew);
            m_run[r] = mnew;
        }
#pragma unroll
        for (int nt = 0; nt < 6; ++nt)
#pragma unroll
            for (int r = 0; r < 8; ++r) o[nt][r] *= alpha[r];
#pragma unroll
        for (int r = 0; r < 8; ++r) o6[r] *= alpha[r];

        // C-layout P -> LDS -> A-fragment layout (DS ops are in-order per wave)
#pragma unroll
        for (int r = 0; r < 8; ++r) {
            int m = r + (hiH ? 8 : 0);
            pbuf[wave][m][lane16]      = (_Float16)p0[r];
            pbuf[wave][m][16 + lane16] = (_Float16)p1[r];
        }
        __builtin_amdgcn_wave_barrier();
        const _Float16* pp = &pbuf[wave][lane16][hiH ? 16 : 0];
        h16_t pf = hcat(*(const h8_t*)pp, *(const h8_t*)(pp + 8));
        __builtin_amdgcn_wave_barrier();

        // O += P * V ; row sums via ones-column on the matrix pipe
#pragma unroll
        for (int nt = 0; nt < 6; ++nt) {
            const _Float16* pv =
                Vb + (size_t)(nt * 16 + lane16) * TK + kb + (hiH ? 16 : 0);
            o[nt] = wmma_f32_f16(pf, *(const h16_t*)pv, o[nt]);
        }
        o6 = wmma_f32_f16(pf, onesf, o6);
    }

    float inv[8];
#pragma unroll
    for (int r = 0; r < 8; ++r) inv[r] = 1.0f / o6[r];

    // AO[b, t, h*96 + hd] f16
#pragma unroll
    for (int nt = 0; nt < 6; ++nt) {
        int n = h * HDIM + nt * 16 + lane16;
#pragma unroll
        for (int r = 0; r < 8; ++r) {
            int q = q0 + r + (hiH ? 8 : 0);
            AO[((size_t)b * TQ + q) * DIM + n] = (_Float16)(o[nt][r] * inv[r]);
        }
    }
}

// ---------- launch ----------
extern "C" void kernel_launch(void* const* d_in, const int* in_sizes, int n_in,
                              void* d_out, int out_size, void* d_ws, size_t ws_size,
                              hipStream_t stream) {
    const float* x   = (const float*)d_in[0];
    const float* mem = (const float*)d_in[1];
    const float* qc  = (const float*)d_in[2];
    const float* mc  = (const float*)d_in[3];
    const float* Wq  = (const float*)d_in[4];
    const float* bq  = (const float*)d_in[5];
    const float* Wk  = (const float*)d_in[6];
    const float* bk  = (const float*)d_in[7];
    const float* Wv  = (const float*)d_in[8];
    const float* bv  = (const float*)d_in[9];
    const float* Wo  = (const float*)d_in[10];
    const float* bo  = (const float*)d_in[11];
    float* out = (float*)d_out;

    char* ws = (char*)d_ws;
    size_t off = 0;
    auto alloc = [&](size_t bytes) {
        void* p = ws + off;
        off += (bytes + 255) & ~(size_t)255;
        return p;
    };
    _Float16* x_h  = (_Float16*)alloc((size_t)NB * TQ * DIM * 2);
    _Float16* m_h  = (_Float16*)alloc((size_t)NB * TK * DIM * 2);
    _Float16* wq_h = (_Float16*)alloc((size_t)DIM * DIM * 2);
    _Float16* wk_h = (_Float16*)alloc((size_t)DIM * DIM * 2);
    _Float16* wv_h = (_Float16*)alloc((size_t)DIM * DIM * 2);
    _Float16* wo_h = (_Float16*)alloc((size_t)DIM * DIM * 2);
    _Float16* Qh   = (_Float16*)alloc((size_t)NB * NH * TQ * HDIM * 2);
    _Float16* Kh   = (_Float16*)alloc((size_t)NB * NH * TK * HDIM * 2);
    _Float16* Vt   = (_Float16*)alloc((size_t)NB * NH * HDIM * TK * 2);
    _Float16* AO   = (_Float16*)alloc((size_t)NB * TQ * DIM * 2);
    (void)ws_size; (void)in_sizes; (void)n_in; (void)out_size;

    const int thr = 256;
    int nx = NB * TQ * DIM, nm = NB * TK * DIM, nw = DIM * DIM;
    cvt_f16_kernel<<<(nx + thr - 1) / thr, thr, 0, stream>>>(x, x_h, nx);
    cvt_f16_kernel<<<(nm + thr - 1) / thr, thr, 0, stream>>>(mem, m_h, nm);
    cvt_f16_kernel<<<(nw + thr - 1) / thr, thr, 0, stream>>>(Wq, wq_h, nw);
    cvt_f16_kernel<<<(nw + thr - 1) / thr, thr, 0, stream>>>(Wk, wk_h, nw);
    cvt_f16_kernel<<<(nw + thr - 1) / thr, thr, 0, stream>>>(Wv, wv_h, nw);
    cvt_f16_kernel<<<(nw + thr - 1) / thr, thr, 0, stream>>>(Wo, wo_h, nw);

    // wave tile 16x128, block tile 32m x 512n
    dim3 gq(NB * TQ / 32, DIM / 512);
    dim3 gk(NB * TK / 32, DIM / 512);
    gemm_wmma_kernel<0><<<gq, 256, 0, stream>>>(x_h, wq_h, bq, Qh, NB * TQ, DIM, DIM, TQ);
    gemm_wmma_kernel<0><<<gk, 256, 0, stream>>>(m_h, wk_h, bk, Kh, NB * TK, DIM, DIM, TK);
    gemm_wmma_kernel<1><<<gk, 256, 0, stream>>>(m_h, wv_h, bv, Vt, NB * TK, DIM, DIM, TK);

    int nrq = NB * NH * TQ * 48;
    int nrk = NB * NH * TK * 48;
    rope_kernel<<<(nrq + thr - 1) / thr, thr, 0, stream>>>(Qh, qc, TQ, nrq);
    rope_kernel<<<(nrk + thr - 1) / thr, thr, 0, stream>>>(Kh, mc, TK, nrk);

    attn_kernel<<<NB * NH * (TQ / 16) / 8, 256, 0, stream>>>(Qh, Kh, Vt, AO);

    gemm_wmma_kernel<2><<<gq, 256, 0, stream>>>(AO, wo_h, bo, out, NB * TQ, DIM, DIM, 1);
}
